// Qwen3SelectiveModel_86612310491526
// MI455X (gfx1250) — matrix operand order repaired and checked by reading, not compiled, not run on hardware
//
#include <hip/hip_runtime.h>
#include <hip/hip_bf16.h>
#include <cstdint>

typedef __attribute__((ext_vector_type(16))) __bf16 v16bf;
typedef __attribute__((ext_vector_type(8)))  __bf16 v8bf;
typedef __attribute__((ext_vector_type(8)))  float  v8f;
typedef __attribute__((ext_vector_type(4)))  uint32_t u32x4;
typedef __attribute__((ext_vector_type(8)))  int32_t  i32x8;
typedef __attribute__((ext_vector_type(4)))  int32_t  i32x4;

static constexpr int TT = 2048;   // tokens
static constexpr int DD = 2048;   // model dim
static constexpr int NH = 16;     // q heads
static constexpr int NKV = 8;     // kv heads
static constexpr int DHD = 128;   // head dim
static constexpr int FFD = 8192;  // ffn dim
static constexpr int NL = 2;      // layers
static constexpr float RMS_EPS = 1e-6f;
static constexpr float ATTN_SCALE = 0.08838834764831845f; // 1/sqrt(128)

__device__ __forceinline__ v8f wmma_bf16(v16bf a, v16bf b, v8f c) {
  return __builtin_amdgcn_wmma_f32_16x16x32_bf16(false, a, false, b, (short)0, c,
                                                 false, false);
}

__device__ __forceinline__ v16bf cat16(v8bf lo, v8bf hi) {
  return __builtin_shufflevector(lo, hi, 0,1,2,3,4,5,6,7,8,9,10,11,12,13,14,15);
}

// Issue one 2D TDM tile load (global row-major -> padded LDS rows).
// tile_d0 elements (bf16) per row, tile_d1 rows, global row stride = k_stride
// elements.  LDS rows are padded: 16 DWORDs data + 4 DWORDs pad (80B stride).
__device__ __forceinline__ void tdm_load_tile(uint32_t lds_off, uint64_t gaddr,
                                              uint32_t k_stride,
                                              uint32_t tile_d0,
                                              uint32_t tile_d1) {
  u32x4 g0;
  g0[0] = 1u;                                  // count=1, user mode, no gather
  g0[1] = lds_off;                             // LDS byte address
  g0[2] = (uint32_t)gaddr;                     // global addr [31:0]
  g0[3] = (uint32_t)(gaddr >> 32) | (2u << 30);// addr [56:32] | type=2 (image)
  uint32_t dim0 = k_stride;                    // tensor extent dim0 (elements)
  uint32_t dim1 = tile_d1;                     // tensor extent dim1 (rows)
  uint64_t stride0 = (uint64_t)k_stride;       // dim0 stride (elements)
  u32x4 g1a;  // group1 words 0..3
  u32x4 g1b;  // group1 words 4..7
  // data_size=1 (2B) | pad_enable | pad_interval=3 (16 DW) | pad_amount=3 (4 DW)
  g1a[0] = (1u << 16) | (1u << 20) | (3u << 22) | (3u << 25);
  g1a[1] = (dim0 & 0xFFFFu) << 16;                       // tensor_dim0 lo16
  g1a[2] = (dim0 >> 16) | ((dim1 & 0xFFFFu) << 16);      // dim0 hi | dim1 lo
  g1a[3] = (dim1 >> 16) | (tile_d0 << 16);               // dim1 hi | tile_dim0
  g1b[0] = tile_d1 & 0xFFFFu;                            // tile_dim1 (dim2=0)
  g1b[1] = (uint32_t)stride0;                            // dim0 stride lo32
  g1b[2] = (uint32_t)(stride0 >> 32);                    // stride hi16 | s1 lo
  g1b[3] = 0u;                                           // stride1 hi
  i32x8 g1;
#pragma unroll
  for (int i = 0; i < 4; ++i) { g1[i] = (int32_t)g1a[i]; g1[i + 4] = (int32_t)g1b[i]; }
  i32x4 z4 = (i32x4){0, 0, 0, 0};
  i32x8 z8 = (i32x8){0, 0, 0, 0, 0, 0, 0, 0};
#if __has_builtin(__builtin_amdgcn_tensor_load_to_lds)
#if __clang_major__ >= 23
  // amdgpu-toolchain (therock-10.0 headers): 6-arg form
  __builtin_amdgcn_tensor_load_to_lds(g0, g1, z4, z4, z8, 0);
#else
  // ROCm 7.2 (clang-22): 5-arg form
  __builtin_amdgcn_tensor_load_to_lds(g0, g1, z4, z4, 0);
#endif
  (void)z8;
#else
  asm volatile("tensor_load_to_lds %0, %1" ::"s"(g0), "s"(g1) : "memory");
  (void)z4;
  (void)z8;
#endif
}

// ---------------------------------------------------------------- f32 -> bf16
__global__ __launch_bounds__(256) void conv_bf16_k(const float* __restrict__ in,
                                                   __bf16* __restrict__ out,
                                                   size_t n) {
  size_t i = (size_t)blockIdx.x * blockDim.x + threadIdx.x;
  size_t stride = (size_t)gridDim.x * blockDim.x;
  for (; i < n; i += stride) out[i] = (__bf16)in[i];
}

// ---------------------------------------------------------------- rope tables
__global__ void rope_tables_k(const int* __restrict__ pos,
                              float* __restrict__ cosb,
                              float* __restrict__ sinb) {
  int t = blockIdx.x;
  int i = threadIdx.x;  // < DHD/2
  float inv = __powf(1.0e6f, -(float)i / (float)(DHD / 2));
  float f = (float)pos[t] * inv;
  cosb[t * (DHD / 2) + i] = cosf(f);
  sinb[t * (DHD / 2) + i] = sinf(f);
}

// ---------------------------------------------------------------- rmsnorm row
template <typename OT>
__global__ __launch_bounds__(256) void rmsnorm_k(const float* __restrict__ x,
                                                 const float* __restrict__ w,
                                                 OT* __restrict__ out, int ncols) {
  int row = blockIdx.x;
  int tid = threadIdx.x;
  const float* xr = x + (size_t)row * ncols;
  float ss = 0.f;
  for (int c = tid; c < ncols; c += 256) {
    float v = xr[c];
    ss += v * v;
  }
  __shared__ float red[256];
  red[tid] = ss;
  __syncthreads();
  for (int s = 128; s > 0; s >>= 1) {
    if (tid < s) red[tid] += red[tid + s];
    __syncthreads();
  }
  float inv = rsqrtf(red[0] / (float)ncols + RMS_EPS);
  OT* orow = out + (size_t)row * ncols;
  for (int c = tid; c < ncols; c += 256) orow[c] = (OT)(xr[c] * inv * w[c]);
}

// ---------------------------------------------------------------- bf16 GEMM
// C[M,N] = A[M,K] @ B[K,N]; block tile 128x128, wave tile 32x64, K-step 32.
// A tile staged by the Tensor Data Mover (padded rows), B tile staged
// transposed by the threads.  MODE 0: f32 out, 1: f32 out + residual,
// 2: bf16 out = silu(aux) * acc  (SwiGLU fusion)
template <int MODE>
__global__ __launch_bounds__(256) void gemm_bf16_k(
    const __bf16* __restrict__ A, const __bf16* __restrict__ Bw,
    float* __restrict__ Cf, __bf16* __restrict__ Cb,
    const float* __restrict__ res, const float* __restrict__ aux,
    int M, int N, int K) {
  __shared__ alignas(64) __bf16 As[128][40];   // 128 x 32, padded (TDM pads)
  __shared__ alignas(64) __bf16 Bst[128][48];  // B tile transposed [n][k]
  int tid = threadIdx.x;
  int wave = tid >> 5, lane = tid & 31;
  int m0 = blockIdx.x * 128, n0 = blockIdx.y * 128;
  int wm = (wave & 3) * 32, wn = (wave >> 2) * 64;
  int colL = lane & 15;
  int akb = (lane & 16) ? 16 : 0;  // A-frag K base (contiguous k = 16*half + s, same as B)
  int bkb = (lane & 16) ? 16 : 0;  // B-frag K base
  v8f acc[2][4];
#pragma unroll
  for (int i = 0; i < 2; ++i)
#pragma unroll
    for (int j = 0; j < 4; ++j)
      acc[i][j] = (v8f){0.f, 0.f, 0.f, 0.f, 0.f, 0.f, 0.f, 0.f};

  uint32_t as_lds =
      (uint32_t)(size_t)(__bf16 __attribute__((address_space(3)))*)&As[0][0];
  uint64_t a_base = (uint64_t)(uintptr_t)(A + (size_t)m0 * K);

  for (int k0 = 0; k0 < K; k0 += 32) {
    // --- A tile via Tensor Data Mover (wave 0 issues for the workgroup) ---
    if (wave == 0) {
      tdm_load_tile(as_lds, a_base + (uint64_t)k0 * 2, (uint32_t)K, 32u, 128u);
    }
    // --- B tile, transposed into LDS by all threads ---
#pragma unroll
    for (int it = 0; it < 2; ++it) {
      int idx = tid + it * 256;
      int kr = idx >> 4, nc = (idx & 15) * 8;
      v8bf bv = *(const v8bf*)(Bw + (size_t)(k0 + kr) * N + n0 + nc);
#pragma unroll
      for (int e = 0; e < 8; ++e) Bst[nc + e][kr] = bv[e];
      if (k0 + 32 < K)
        __builtin_prefetch(Bw + (size_t)(k0 + 32 + kr) * N + n0 + nc, 0, 1);
    }
    if (wave == 0) __builtin_amdgcn_s_wait_tensorcnt(0);
    __syncthreads();
    // --- fragments + 8 WMMAs per wave per K-step ---
    v16bf af[2];
#pragma unroll
    for (int i = 0; i < 2; ++i) {
      const __bf16* ap = &As[wm + i * 16 + colL][0];
      v8bf lo = *(const v8bf*)(ap + akb);
      v8bf hi = *(const v8bf*)(ap + akb + 8);
      af[i] = cat16(lo, hi);
    }
#pragma unroll
    for (int j = 0; j < 4; ++j) {
      v16bf bf = *(const v16bf*)&Bst[wn + j * 16 + colL][bkb];
#pragma unroll
      for (int i = 0; i < 2; ++i) acc[i][j] = wmma_bf16(af[i], bf, acc[i][j]);
    }
    __syncthreads();
  }
  int rb = (lane & 16) ? 8 : 0;
#pragma unroll
  for (int i = 0; i < 2; ++i) {
#pragma unroll
    for (int j = 0; j < 4; ++j) {
      int n = n0 + wn + j * 16 + colL;
#pragma unroll
      for (int r = 0; r < 8; ++r) {
        int m = m0 + wm + i * 16 + rb + r;
        size_t idx = (size_t)m * N + n;
        float v = acc[i][j][r];
        if constexpr (MODE == 0) {
          Cf[idx] = v;
        } else if constexpr (MODE == 1) {
          Cf[idx] = res[idx] + v;
        } else {
          float g = aux[idx];
          float s = g / (1.f + __expf(-g));
          Cb[idx] = (__bf16)(s * v);
        }
      }
    }
  }
}

// ------------------------------------------------- q/k norm + rope, v transpose
__global__ __launch_bounds__(128) void qkv_post_k(
    const float* __restrict__ qf, const float* __restrict__ kf,
    const float* __restrict__ vf, const float* __restrict__ qn_w,
    const float* __restrict__ kn_w, const float* __restrict__ cosb,
    const float* __restrict__ sinb, __bf16* __restrict__ qb,
    __bf16* __restrict__ kbb, __bf16* __restrict__ vtb) {
  int t = blockIdx.x;
  int y = blockIdx.y;  // [0,NH): q head; [NH,NH+NKV): k head; else v head
  int d = threadIdx.x; // < DHD
  if (y >= NH + NKV) {
    int hk = y - (NH + NKV);
    float v = vf[(size_t)t * (NKV * DHD) + hk * DHD + d];
    vtb[((size_t)(hk * DHD + d)) * TT + t] = (__bf16)v;
    return;
  }
  const float* src;
  const float* wn;
  __bf16* dst;
  if (y < NH) {
    src = qf + (size_t)t * (NH * DHD) + y * DHD;
    wn = qn_w;
    dst = qb + (size_t)t * (NH * DHD) + y * DHD;
  } else {
    int hk = y - NH;
    src = kf + (size_t)t * (NKV * DHD) + hk * DHD;
    wn = kn_w;
    dst = kbb + (size_t)t * (NKV * DHD) + hk * DHD;
  }
  float v = src[d];
  __shared__ float red[128];
  red[d] = v * v;
  __syncthreads();
  for (int s = 64; s > 0; s >>= 1) {
    if (d < s) red[d] += red[d + s];
    __syncthreads();
  }
  float inv = rsqrtf(red[0] / (float)DHD + RMS_EPS);
  float n = v * inv * wn[d];
  int p = d ^ 64;
  float np = src[p] * inv * wn[p];
  float cs = cosb[t * (DHD / 2) + (d & 63)];
  float sn = sinb[t * (DHD / 2) + (d & 63)];
  float rot = (d < 64) ? -np : np;
  dst[d] = (__bf16)(n * cs + rot * sn);
}

// ---------------------------------------------------------------- flash attn
// one wave per (head, 16-query tile); 32-key chunks; online softmax
__global__ __launch_bounds__(256) void attn_k(
    const __bf16* __restrict__ qb, const __bf16* __restrict__ kbb,
    const __bf16* __restrict__ vtb, const int* __restrict__ seg,
    const int* __restrict__ pos, __bf16* __restrict__ ob) {
  __shared__ alignas(64) __bf16 Pl[8][16][40];  // per-wave P staging
  int wave = threadIdx.x >> 5;
  int lane = threadIdx.x & 31;
  int gw = blockIdx.x * 8 + wave;
  int nqt = TT / 16;
  int qtile = gw % nqt;
  int h = gw / nqt;
  int kvh = h / (NH / NKV);
  int qt0 = qtile * 16;
  int colL = lane & 15;
  int rbase = (lane & 16) ? 8 : 0;
  int akb = (lane & 16) ? 16 : 0;
  int bkb = (lane & 16) ? 16 : 0;

  // Q A-fragments over DH=128 (4 x K32)
  v16bf qa[4];
  {
    const __bf16* qp = qb + (size_t)(qt0 + colL) * (NH * DHD) + h * DHD;
#pragma unroll
    for (int kk = 0; kk < 4; ++kk) {
      v8bf lo = *(const v8bf*)(qp + kk * 32 + akb);
      v8bf hi = *(const v8bf*)(qp + kk * 32 + akb + 8);
      qa[kk] = cat16(lo, hi);
    }
  }
  int sq[8];
#pragma unroll
  for (int r = 0; r < 8; ++r) sq[r] = seg[qt0 + rbase + r];
  float mrow[8], lrow[8];
#pragma unroll
  for (int r = 0; r < 8; ++r) { mrow[r] = -1e30f; lrow[r] = 0.f; }
  v8f oacc[8];
#pragma unroll
  for (int dt = 0; dt < 8; ++dt)
    oacc[dt] = (v8f){0.f, 0.f, 0.f, 0.f, 0.f, 0.f, 0.f, 0.f};

  int ks = (qt0 - pos[qt0]) & ~31;  // span start, 32-aligned down
  for (int kt0 = ks; kt0 <= qt0 + 15; kt0 += 32) {
    v8f s0 = (v8f){0.f, 0.f, 0.f, 0.f, 0.f, 0.f, 0.f, 0.f};
    v8f s1 = (v8f){0.f, 0.f, 0.f, 0.f, 0.f, 0.f, 0.f, 0.f};
    int ktA = kt0 + colL;
    int ktB = kt0 + 16 + colL;
    {
      const __bf16* kp0 =
          kbb + (size_t)min(ktA, TT - 1) * (NKV * DHD) + kvh * DHD + bkb;
      const __bf16* kp1 =
          kbb + (size_t)min(ktB, TT - 1) * (NKV * DHD) + kvh * DHD + bkb;
#pragma unroll
      for (int kk = 0; kk < 4; ++kk)
        s0 = wmma_bf16(qa[kk], *(const v16bf*)(kp0 + kk * 32), s0);
#pragma unroll
      for (int kk = 0; kk < 4; ++kk)
        s1 = wmma_bf16(qa[kk], *(const v16bf*)(kp1 + kk * 32), s1);
    }
    int sk0 = seg[min(ktA, TT - 1)];
    int sk1 = seg[min(ktB, TT - 1)];
    float alpha[8];
#pragma unroll
    for (int r = 0; r < 8; ++r) {
      int qt = qt0 + rbase + r;
      float a0 = s0[r] * ATTN_SCALE;
      float a1 = s1[r] * ATTN_SCALE;
      bool ok0 = (ktA <= qt) && (sk0 == sq[r]);
      bool ok1 = (ktB <= qt) && (sk1 == sq[r]);
      a0 = ok0 ? a0 : -1e30f;
      a1 = ok1 ? a1 : -1e30f;
      float mx = fmaxf(a0, a1);
#pragma unroll
      for (int off = 1; off < 16; off <<= 1)
        mx = fmaxf(mx, __shfl_xor(mx, off, 32));
      float mn = fmaxf(mrow[r], mx);
      float p0 = ok0 ? __expf(a0 - mn) : 0.f;
      float p1 = ok1 ? __expf(a1 - mn) : 0.f;
      Pl[wave][rbase + r][colL] = (__bf16)p0;
      Pl[wave][rbase + r][16 + colL] = (__bf16)p1;
      float rsum = p0 + p1;
#pragma unroll
      for (int off = 1; off < 16; off <<= 1) rsum += __shfl_xor(rsum, off, 32);
      float al = __expf(mrow[r] - mn);
      lrow[r] = lrow[r] * al + rsum;
      mrow[r] = mn;
      alpha[r] = al;
    }
#pragma unroll
    for (int dt = 0; dt < 8; ++dt)
#pragma unroll
      for (int r = 0; r < 8; ++r) oacc[dt][r] *= alpha[r];
    // P as A-fragment (relayout through LDS, same-wave DS ordering)
    const __bf16* pp = &Pl[wave][colL][0];
    v8bf plo = *(const v8bf*)(pp + akb);
    v8bf phi = *(const v8bf*)(pp + akb + 8);
    v16bf pa = cat16(plo, phi);
#pragma unroll
    for (int dt = 0; dt < 8; ++dt) {
      const __bf16* vp =
          vtb + (size_t)(kvh * DHD + dt * 16 + colL) * TT + kt0 + bkb;
      oacc[dt] = wmma_bf16(pa, *(const v16bf*)vp, oacc[dt]);
    }
  }
#pragma unroll
  for (int dt = 0; dt < 8; ++dt) {
#pragma unroll
    for (int r = 0; r < 8; ++r) {
      float l = lrow[r];
      float o = (l > 0.f) ? oacc[dt][r] / l : 0.f;
      ob[(size_t)(qt0 + rbase + r) * (NH * DHD) + h * DHD + dt * 16 + colL] =
          (__bf16)o;
    }
  }
}

// ---------------------------------------------------------------- launcher
extern "C" void kernel_launch(void* const* d_in, const int* in_sizes, int n_in,
                              void* d_out, int out_size, void* d_ws,
                              size_t ws_size, hipStream_t stream) {
  const float* hidden = (const float*)d_in[0];
  const float* ln1 = (const float*)d_in[1];
  const float* ln2 = (const float*)d_in[2];
  const float* wq = (const float*)d_in[3];
  const float* wk = (const float*)d_in[4];
  const float* wv = (const float*)d_in[5];
  const float* wo = (const float*)d_in[6];
  const float* qnw = (const float*)d_in[7];
  const float* knw = (const float*)d_in[8];
  const float* wg = (const float*)d_in[9];
  const float* wu = (const float*)d_in[10];
  const float* wd = (const float*)d_in[11];
  const float* fnw = (const float*)d_in[12];
  const int* seg = (const int*)d_in[13];
  const int* pos = (const int*)d_in[14];
  float* out = (float*)d_out;

  char* p = (char*)d_ws;
  auto alloc = [&](size_t bytes) {
    char* r = p;
    p += (bytes + 255) & ~size_t(255);
    return r;
  };
  float* x = (float*)alloc((size_t)TT * DD * 4);
  __bf16* hb = (__bf16*)alloc((size_t)TT * DD * 2);
  float* qf = (float*)alloc((size_t)TT * NH * DHD * 4);
  float* kf = (float*)alloc((size_t)TT * NKV * DHD * 4);
  float* vf = (float*)alloc((size_t)TT * NKV * DHD * 4);
  __bf16* qb = (__bf16*)alloc((size_t)TT * NH * DHD * 2);
  __bf16* kbb = (__bf16*)alloc((size_t)TT * NKV * DHD * 2);
  __bf16* vtb = (__bf16*)alloc(((size_t)NKV * DHD * TT + 64) * 2);
  __bf16* ob = (__bf16*)alloc((size_t)TT * NH * DHD * 2);
  float* gbuf = (float*)alloc((size_t)TT * FFD * 4);
  __bf16* mlpa = (__bf16*)alloc((size_t)TT * FFD * 2);
  float* cosb = (float*)alloc((size_t)TT * (DHD / 2) * 4);
  float* sinb = (float*)alloc((size_t)TT * (DHD / 2) * 4);
  __bf16* wcv = (__bf16*)alloc((size_t)DD * FFD * 2);  // largest weight slice

  (void)hipMemcpyAsync(x, hidden, (size_t)TT * DD * 4,
                       hipMemcpyDeviceToDevice, stream);
  rope_tables_k<<<TT, DHD / 2, 0, stream>>>(pos, cosb, sinb);

  auto conv = [&](const float* w, size_t n) {
    conv_bf16_k<<<2048, 256, 0, stream>>>(w, wcv, n);
  };
  dim3 blk(256);
  for (int l = 0; l < NL; ++l) {
    rmsnorm_k<__bf16><<<TT, 256, 0, stream>>>(x, ln1 + (size_t)l * DD, hb, DD);
    // QKV projections
    conv(wq + (size_t)l * DD * NH * DHD, (size_t)DD * NH * DHD);
    gemm_bf16_k<0><<<dim3(TT / 128, (NH * DHD) / 128), blk, 0, stream>>>(
        hb, wcv, qf, nullptr, nullptr, nullptr, TT, NH * DHD, DD);
    conv(wk + (size_t)l * DD * NKV * DHD, (size_t)DD * NKV * DHD);
    gemm_bf16_k<0><<<dim3(TT / 128, (NKV * DHD) / 128), blk, 0, stream>>>(
        hb, wcv, kf, nullptr, nullptr, nullptr, TT, NKV * DHD, DD);
    conv(wv + (size_t)l * DD * NKV * DHD, (size_t)DD * NKV * DHD);
    gemm_bf16_k<0><<<dim3(TT / 128, (NKV * DHD) / 128), blk, 0, stream>>>(
        hb, wcv, vf, nullptr, nullptr, nullptr, TT, NKV * DHD, DD);
    // per-head rmsnorm + rope + v transpose
    qkv_post_k<<<dim3(TT, NH + 2 * NKV), 128, 0, stream>>>(
        qf, kf, vf, qnw + (size_t)l * DHD, knw + (size_t)l * DHD, cosb, sinb,
        qb, kbb, vtb);
    // flash attention
    attn_k<<<(NH * (TT / 16)) / 8, 256, 0, stream>>>(qb, kbb, vtb, seg, pos,
                                                     ob);
    // output projection + residual
    conv(wo + (size_t)l * NH * DHD * DD, (size_t)NH * DHD * DD);
    gemm_bf16_k<1><<<dim3(TT / 128, DD / 128), blk, 0, stream>>>(
        ob, wcv, x, nullptr, x, nullptr, TT, DD, NH * DHD);
    // MLP
    rmsnorm_k<__bf16><<<TT, 256, 0, stream>>>(x, ln2 + (size_t)l * DD, hb, DD);
    conv(wg + (size_t)l * DD * FFD, (size_t)DD * FFD);
    gemm_bf16_k<0><<<dim3(TT / 128, FFD / 128), blk, 0, stream>>>(
        hb, wcv, gbuf, nullptr, nullptr, nullptr, TT, FFD, DD);
    conv(wu + (size_t)l * DD * FFD, (size_t)DD * FFD);
    gemm_bf16_k<2><<<dim3(TT / 128, FFD / 128), blk, 0, stream>>>(
        hb, wcv, nullptr, mlpa, nullptr, gbuf, TT, FFD, DD);
    conv(wd + (size_t)l * FFD * DD, (size_t)FFD * DD);
    gemm_bf16_k<1><<<dim3(TT / 128, DD / 128), blk, 0, stream>>>(
        mlpa, wcv, x, nullptr, x, nullptr, TT, DD, FFD);
  }
  rmsnorm_k<float><<<TT, 256, 0, stream>>>(x, fnw, out, DD);
}
